// LNO2d_13391708029805
// MI455X (gfx1250) — matrix operand order repaired and checked by reading, not compiled, not run on hardware
//
#include <hip/hip_runtime.h>

// LNO2d forward for MI455X (gfx1250, wave32, WMMA).
// All matmul-shaped work (2D DFT/iDFT as dense 256x256 symmetric-DFT-matrix
// GEMMs, 1x1 conv, fc1/fc2 head) goes through v_wmma_f32_16x16x32_f16 with
// f32 accumulation. LDS tiles are laid out so every lane's WMMA fragment is
// two contiguous 16B runs -> ds_load_b128 pairs. The bandwidth-dominant DFT
// GEMMs use an unguarded full-tile kernel with float4 global loads.

typedef __attribute__((ext_vector_type(16))) _Float16 v16h;
typedef __attribute__((ext_vector_type(8)))  _Float16 v8h_t;
typedef __attribute__((ext_vector_type(4)))  _Float16 v4h_t;
typedef __attribute__((ext_vector_type(8)))  float    v8f;

#define PI_F 3.14159265358979323846f
#define SHUF16(lo, hi) __builtin_shufflevector(lo, hi, 0,1,2,3,4,5,6,7,8,9,10,11,12,13,14,15)

// ---------------------------------------------------------------------------
// FULL-TILE batched GEMM: C = alpha*A*B + beta*C, row-major f32.
// Requires: M%16==0, N%256==0, K%32==0, lda%4==0, ldb%4==0, 16B-aligned bases.
// Block: 128 threads = 4 waves; wave computes a 16x64 strip (4 WMMA acc).
// ---------------------------------------------------------------------------
__global__ __launch_bounds__(128)
void wmma_gemm_full(const float* __restrict__ A, const float* __restrict__ B,
                    float* __restrict__ C,
                    int K, int lda, int ldb, int ldc,
                    long sA, long sB, long sC, float alpha, float beta)
{
    const int lane = threadIdx.x, wave = threadIdx.y, tid = wave * 32 + lane;
    const int m0  = blockIdx.y * 16;
    const int n0w = (blockIdx.x * 4 + wave) * 64;

    const float* Ab = A + (long)blockIdx.z * sA;
    const float* Bb = B + (long)blockIdx.z * sB;
    float*       Cb = C + (long)blockIdx.z * sC;

    __shared__ __align__(16) _Float16 As[16 * 32];        // [m][k]
    __shared__ __align__(16) _Float16 Bs[4][64 * 40];     // [n][k], stride 40 halves

    v8f acc0 = {}, acc1 = {}, acc2 = {}, acc3 = {};

    const int ar  = tid >> 3;           // A tile row 0..15
    const int ac  = (tid & 7) * 4;      // A tile col group
    const int brh = lane >> 4;          // 0/1
    const int bc4 = (lane & 15) * 4;    // B col group within wave strip

    const int mA = lane & 15, kb  = (lane >> 4) * 16;
    const int nB = lane & 15, kb2 = (lane >> 4) * 16;

    for (int k0 = 0; k0 < K; k0 += 32) {
        // stage A: one float4 per thread
        {
            const float4 va = *(const float4*)(Ab + (long)(m0 + ar) * lda + k0 + ac);
            v4h_t h;
            h[0] = (_Float16)va.x; h[1] = (_Float16)va.y;
            h[2] = (_Float16)va.z; h[3] = (_Float16)va.w;
            *(v4h_t*)(&As[ar * 32 + ac]) = h;
        }
        // stage B transposed into [n][k]: 16 float4 per lane
#pragma unroll
        for (int t = 0; t < 16; ++t) {
            int r = t * 2 + brh;                           // k row 0..31
            const float4 vb = *(const float4*)(Bb + (long)(k0 + r) * ldb + n0w + bc4);
            Bs[wave][(bc4 + 0) * 40 + r] = (_Float16)vb.x;
            Bs[wave][(bc4 + 1) * 40 + r] = (_Float16)vb.y;
            Bs[wave][(bc4 + 2) * 40 + r] = (_Float16)vb.z;
            Bs[wave][(bc4 + 3) * 40 + r] = (_Float16)vb.w;
        }
        __syncthreads();

        if (k0 + 32 < K) {   // cache hints for next K block (global_prefetch_b8)
            __builtin_prefetch(Ab + (long)(m0 + ar) * lda + k0 + 32 + ac, 0, 1);
            __builtin_prefetch(Bb + (long)(k0 + 32 + brh) * ldb + n0w + bc4, 0, 1);
        }

        // A fragment: two contiguous 16B runs -> ds_load_b128 x2
        v8h_t alo = *(const v8h_t*)(&As[mA * 32 + kb]);
        v8h_t ahi = *(const v8h_t*)(&As[mA * 32 + kb + 8]);
        v16h  af  = SHUF16(alo, ahi);

#pragma unroll
        for (int j = 0; j < 4; ++j) {
            const _Float16* bp = &Bs[wave][(j * 16 + nB) * 40 + kb2];
            v8h_t blo = *(const v8h_t*)(bp);
            v8h_t bhi = *(const v8h_t*)(bp + 8);
            v16h  bf  = SHUF16(blo, bhi);
            v8f&  acc = (j == 0 ? acc0 : j == 1 ? acc1 : j == 2 ? acc2 : acc3);
            acc = __builtin_amdgcn_wmma_f32_16x16x32_f16(false, af, false, bf,
                                                         (short)0, acc, false, false);
        }
        __syncthreads();
    }

    const int nC0   = n0w + (lane & 15);
    const int mBase = m0 + (lane >> 4) * 8;
#pragma unroll
    for (int j = 0; j < 4; ++j) {
        v8f acc = (j == 0 ? acc0 : j == 1 ? acc1 : j == 2 ? acc2 : acc3);
        int nC = nC0 + j * 16;
#pragma unroll
        for (int r = 0; r < 8; ++r) {
            long idx = (long)(mBase + r) * ldc + nC;
            float outv = alpha * acc[r];
            if (beta != 0.f) outv += beta * Cb[idx];
            Cb[idx] = outv;
        }
    }
}

// ---------------------------------------------------------------------------
// GUARDED batched GEMM (any M/N/K, optional transA). Same tile shape.
// Used only for the small edge GEMMs (w0: K=16, fc1: K=16, fc2: M=3).
// ---------------------------------------------------------------------------
__global__ __launch_bounds__(128)
void wmma_gemm_guard(const float* __restrict__ A, const float* __restrict__ B,
                     float* __restrict__ C,
                     int M, int N, int K, int lda, int ldb, int ldc,
                     long sA, long sB, long sC,
                     float alpha, float beta, int transA)
{
    const int lane = threadIdx.x, wave = threadIdx.y, tid = wave * 32 + lane;
    const int m0  = blockIdx.y * 16;
    const int n0w = (blockIdx.x * 4 + wave) * 64;

    const float* Ab = A + (long)blockIdx.z * sA;
    const float* Bb = B + (long)blockIdx.z * sB;
    float*       Cb = C + (long)blockIdx.z * sC;

    __shared__ __align__(16) _Float16 As[16 * 32];
    __shared__ __align__(16) _Float16 Bs[4][64 * 40];

    v8f acc0 = {}, acc1 = {}, acc2 = {}, acc3 = {};

    const int mA = lane & 15, kb  = (lane >> 4) * 16;
    const int nB = lane & 15, kb2 = (lane >> 4) * 16;

    for (int k0 = 0; k0 < K; k0 += 32) {
        // stage A (scalar, guarded): 4 elements per thread
#pragma unroll
        for (int t = 0; t < 4; ++t) {
            int idx = tid * 4 + t;
            int r = idx >> 5, c = idx & 31;
            int gm = m0 + r, gk = k0 + c;
            float val = 0.f;
            if (gm < M && gk < K)
                val = transA ? Ab[(long)gk * lda + gm] : Ab[(long)gm * lda + gk];
            As[idx] = (_Float16)val;
        }
        // stage B (scalar, guarded) into transposed [n][k]: 64 per lane
#pragma unroll
        for (int t = 0; t < 64; ++t) {
            int idx = t * 32 + lane;
            int n = idx >> 5, r = idx & 31;
            int gk = k0 + r, gn = n0w + n;
            float val = 0.f;
            if (gk < K && gn < N)
                val = Bb[(long)gk * ldb + gn];
            Bs[wave][n * 40 + r] = (_Float16)val;
        }
        __syncthreads();

        v8h_t alo = *(const v8h_t*)(&As[mA * 32 + kb]);
        v8h_t ahi = *(const v8h_t*)(&As[mA * 32 + kb + 8]);
        v16h  af  = SHUF16(alo, ahi);
#pragma unroll
        for (int j = 0; j < 4; ++j) {
            const _Float16* bp = &Bs[wave][(j * 16 + nB) * 40 + kb2];
            v8h_t blo = *(const v8h_t*)(bp);
            v8h_t bhi = *(const v8h_t*)(bp + 8);
            v16h  bf  = SHUF16(blo, bhi);
            v8f&  acc = (j == 0 ? acc0 : j == 1 ? acc1 : j == 2 ? acc2 : acc3);
            acc = __builtin_amdgcn_wmma_f32_16x16x32_f16(false, af, false, bf,
                                                         (short)0, acc, false, false);
        }
        __syncthreads();
    }

    const int nC0   = n0w + (lane & 15);
    const int mBase = m0 + (lane >> 4) * 8;
#pragma unroll
    for (int j = 0; j < 4; ++j) {
        v8f acc = (j == 0 ? acc0 : j == 1 ? acc1 : j == 2 ? acc2 : acc3);
        int nC = nC0 + j * 16;
        if (nC >= N) continue;
#pragma unroll
        for (int r = 0; r < 8; ++r) {
            int mC = mBase + r;
            if (mC < M) {
                long idx = (long)mC * ldc + nC;
                float outv = alpha * acc[r];
                if (beta != 0.f) outv += beta * Cb[idx];
                Cb[idx] = outv;
            }
        }
    }
}

// ---------------------------------------------------------------------------
// Lift: v[b,c,h,w] = fc0([x(3), h/255, w/255]) ,  c in 0..15
// ---------------------------------------------------------------------------
__global__ void lift_kernel(const float* __restrict__ x, const float* __restrict__ w,
                            const float* __restrict__ bias, float* __restrict__ v)
{
    long idx = (long)blockIdx.x * blockDim.x + threadIdx.x;   // B*H*W = 524288
    if (idx >= (long)8 * 65536) return;
    int wi = idx & 255, h = (int)((idx >> 8) & 255), b = (int)(idx >> 16);
    long pix = ((long)h << 8) + wi;
    float x0 = x[(((long)(b * 3 + 0)) << 16) + pix];
    float x1 = x[(((long)(b * 3 + 1)) << 16) + pix];
    float x2 = x[(((long)(b * 3 + 2)) << 16) + pix];
    float gx = (float)h  * (1.f / 255.f);
    float gy = (float)wi * (1.f / 255.f);
#pragma unroll
    for (int c = 0; c < 16; ++c) {
        float a = bias[c] + x0 * w[0 * 16 + c] + x1 * w[1 * 16 + c] + x2 * w[2 * 16 + c]
                + gx * w[3 * 16 + c] + gy * w[4 * 16 + c];
        v[(((long)(b * 16 + c)) << 16) + pix] = a;
    }
}

// ---------------------------------------------------------------------------
// InstanceNorm2d (affine=False): one block per (b,c) plane of HW elements.
// ---------------------------------------------------------------------------
__global__ __launch_bounds__(256)
void inorm_kernel(const float* __restrict__ src, float* __restrict__ dst, int HW)
{
    __shared__ float ssum[256], ssq[256];
    const long base = (long)blockIdx.x * HW;
    float s = 0.f, s2 = 0.f;
    for (int i = threadIdx.x; i < HW; i += 256) {
        float v = src[base + i];
        s += v; s2 += v * v;
    }
    ssum[threadIdx.x] = s; ssq[threadIdx.x] = s2;
    __syncthreads();
    for (int off = 128; off > 0; off >>= 1) {
        if ((int)threadIdx.x < off) {
            ssum[threadIdx.x] += ssum[threadIdx.x + off];
            ssq[threadIdx.x]  += ssq[threadIdx.x + off];
        }
        __syncthreads();
    }
    float mu  = ssum[0] / (float)HW;
    float var = ssq[0] / (float)HW - mu * mu;
    float rs  = rsqrtf(var + 1e-5f);
    for (int i = threadIdx.x; i < HW; i += 256)
        dst[base + i] = (src[base + i] - mu) * rs;
}

// 256-point DFT matrix (symmetric): F[j,k] = exp(-2*pi*i*j*k/256)
__global__ void dftmat_kernel(float* __restrict__ Fr, float* __restrict__ Fi)
{
    int idx = blockIdx.x * blockDim.x + threadIdx.x;
    if (idx >= 65536) return;
    int j = idx >> 8, k = idx & 255;
    int m = (j * k) & 255;
    float ang = -2.f * PI_F * (float)m / 256.f;
    Fr[idx] = cosf(ang);
    Fi[idx] = sinf(ang);
}

// Tables: A1 = 1/(i*w1 - p1), A2 = 1/(i*w2 - p2), E1 = exp(p1*ty), E2 = exp(p2*tx)
__global__ void tables_kernel(const float* __restrict__ p1, const float* __restrict__ p2,
                              const float* __restrict__ tx, const float* __restrict__ ty,
                              float* A1r, float* A1i, float* A2r, float* A2i,
                              float* E1r, float* E1i, float* E2r, float* E2i)
{
    int idx = blockIdx.x * blockDim.x + threadIdx.x;   // 524288
    if (idx >= 524288) return;
    int n = idx & 255, m = (idx >> 8) & 7, k = (idx >> 11) & 15, i = idx >> 15;
    int kint = (n < 128) ? n : n - 256;
    float dty = ty[1] - ty[0];
    float dtx = tx[1] - tx[0];
    int pidx = (((i * 16 + k) * 8) + m) * 2;
    {
        float pr = p1[pidx], pim = p1[pidx + 1];
        float w1 = 2.f * PI_F * (float)kint / (256.f * dty);
        float dr = -pr, di = w1 - pim;
        float inv = 1.f / (dr * dr + di * di);
        A1r[idx] = dr * inv;  A1i[idx] = -di * inv;
        float t = ty[n], ea = expf(pr * t);
        E1r[idx] = ea * cosf(pim * t);  E1i[idx] = ea * sinf(pim * t);
    }
    {
        float pr = p2[pidx], pim = p2[pidx + 1];
        float w2 = 2.f * PI_F * (float)kint / (256.f * dtx);
        float dr = -pr, di = w2 - pim;
        float inv = 1.f / (dr * dr + di * di);
        A2r[idx] = dr * inv;  A2i[idx] = -di * inv;
        float t = tx[n], ea = expf(pr * t);
        E2r[idx] = ea * cosf(pim * t);  E2i[idx] = ea * sinf(pim * t);
    }
}

// T[i,k,q,o] = sum_p res[i,k,p,q] * A1[i,k,p,o]
__global__ void ttab_kernel(const float* __restrict__ res,
                            const float* __restrict__ A1r, const float* __restrict__ A1i,
                            float* __restrict__ Ttr, float* __restrict__ Tti)
{
    int idx = blockIdx.x * blockDim.x + threadIdx.x;   // 524288
    if (idx >= 524288) return;
    int o = idx & 255, q = (idx >> 8) & 7, k = (idx >> 11) & 15, i = idx >> 15;
    float sr = 0.f, si = 0.f;
#pragma unroll
    for (int p = 0; p < 8; ++p) {
        int ridx = ((((i * 16 + k) * 8 + p) * 8) + q) * 2;
        float rr = res[ridx], ri = res[ridx + 1];
        int a = (((i * 16 + k) * 8 + p) << 8) + o;
        float ar = A1r[a], ai = A1i[a];
        sr += rr * ar - ri * ai;
        si += rr * ai + ri * ar;
    }
    Ttr[idx] = sr; Tti[idx] = si;
}

// O[b,k,o,x] = sum_i alpha[b,i,o,x]*G[i,k,o,x], G = sum_q T[i,k,q,o]*A2[i,k,q,x]
__global__ __launch_bounds__(256)
void okernel(const float* __restrict__ Ar, const float* __restrict__ Ai,
             const float* __restrict__ Ttr, const float* __restrict__ Tti,
             const float* __restrict__ A2r, const float* __restrict__ A2i,
             float* __restrict__ Or, float* __restrict__ Oi)
{
    long idx = (long)blockIdx.x * blockDim.x + threadIdx.x;   // 16*65536
    if (idx >= (long)16 * 65536) return;
    int x = (int)(idx & 255), o = (int)((idx >> 8) & 255), k = (int)(idx >> 16);
    float Gr[16], Gi[16];
#pragma unroll 4
    for (int i = 0; i < 16; ++i) {
        float sr = 0.f, si = 0.f;
#pragma unroll
        for (int q = 0; q < 8; ++q) {
            int tIdx = (((i * 16 + k) * 8 + q) << 8) + o;
            int aIdx = (((i * 16 + k) * 8 + q) << 8) + x;
            float tr = Ttr[tIdx], ti = Tti[tIdx];
            float ar = A2r[aIdx], ai = A2i[aIdx];
            sr += tr * ar - ti * ai;
            si += tr * ai + ti * ar;
        }
        Gr[i] = sr; Gi[i] = si;
    }
    long px = ((long)o << 8) + x;
    for (int b = 0; b < 8; ++b) {
        float sr = 0.f, si = 0.f;
#pragma unroll 4
        for (int i = 0; i < 16; ++i) {
            long aIdx = (((long)(b * 16 + i)) << 16) + px;
            float ar = Ar[aIdx], ai = Ai[aIdx];
            sr += ar * Gr[i] - ai * Gi[i];
            si += ar * Gi[i] + ai * Gr[i];
        }
        long oIdx = (((long)(b * 16 + k)) << 16) + px;
        Or[oIdx] = sr; Oi[oIdx] = si;
    }
}

// S[b,i,k,o,q] = sum_x alpha[b,i,o,x] * A2[i,k,q,x]
__global__ __launch_bounds__(256)
void skernel(const float* __restrict__ Ar, const float* __restrict__ Ai,
             const float* __restrict__ A2r, const float* __restrict__ A2i,
             float* __restrict__ Sr, float* __restrict__ Si)
{
    long idx = (long)blockIdx.x * blockDim.x + threadIdx.x;   // 4,194,304
    if (idx >= (long)4194304) return;
    int q = (int)(idx & 7), o = (int)((idx >> 3) & 255);
    int k = (int)((idx >> 11) & 15), i = (int)((idx >> 15) & 15), b = (int)(idx >> 19);
    const float* ar = Ar + (((long)(b * 16 + i)) << 16) + ((long)o << 8);
    const float* ai = Ai + (((long)(b * 16 + i)) << 16) + ((long)o << 8);
    const float* br = A2r + ((((i * 16 + k) * 8 + q)) << 8);
    const float* bi = A2i + ((((i * 16 + k) * 8 + q)) << 8);
    float sr = 0.f, si = 0.f;
    for (int x = 0; x < 256; ++x) {
        float r1 = ar[x], i1 = ai[x];
        float r2 = br[x], i2 = bi[x];
        sr += r1 * r2 - i1 * i2;
        si += r1 * i2 + i1 * r2;
    }
    Sr[idx] = sr; Si[idx] = si;
}

// out2[b,k,p,q] = sum_i ( sum_o S[b,i,k,o,q]*A1[i,k,p,o] ) * res[i,k,p,q]
__global__ void out2_kernel(const float* __restrict__ Sr, const float* __restrict__ Si,
                            const float* __restrict__ A1r, const float* __restrict__ A1i,
                            const float* __restrict__ res,
                            float* __restrict__ o2r, float* __restrict__ o2i)
{
    int idx = blockIdx.x * blockDim.x + threadIdx.x;   // 8192
    if (idx >= 8192) return;
    int q = idx & 7, p = (idx >> 3) & 7, k = (idx >> 6) & 15, b = idx >> 10;
    float accr = 0.f, acci = 0.f;
    for (int i = 0; i < 16; ++i) {
        float mr = 0.f, mi = 0.f;
        int  a1base = (((i * 16 + k) * 8 + p) << 8);
        long sbase  = (((long)((b * 16 + i) * 16 + k)) << 11) + q;
        for (int o = 0; o < 256; ++o) {
            float sr = Sr[sbase + ((long)o << 3)], si = Si[sbase + ((long)o << 3)];
            float ar = A1r[a1base + o], ai = A1i[a1base + o];
            mr += sr * ar - si * ai;
            mi += sr * ai + si * ar;
        }
        int ridx = ((((i * 16 + k) * 8 + p) * 8) + q) * 2;
        float rr = res[ridx], ri = res[ridx + 1];
        accr += mr * rr - mi * ri;
        acci += mr * ri + mi * rr;
    }
    o2r[idx] = accr; o2i[idx] = acci;     // layout ((b*16+k)*8+p)*8+q
}

// U[n,b2,i2,q,z] = sum_p out2[n,b2,p,q] * E1[b2,i2,p,z]
__global__ void ukernel(const float* __restrict__ o2r, const float* __restrict__ o2i,
                        const float* __restrict__ E1r, const float* __restrict__ E1i,
                        float* __restrict__ Ur, float* __restrict__ Ui)
{
    long idx = (long)blockIdx.x * blockDim.x + threadIdx.x;   // 4,194,304
    if (idx >= (long)4194304) return;
    int z = (int)(idx & 255), q = (int)((idx >> 8) & 7);
    int i2 = (int)((idx >> 11) & 15), b2 = (int)((idx >> 15) & 15), n = (int)(idx >> 19);
    float sr = 0.f, si = 0.f;
#pragma unroll
    for (int p = 0; p < 8; ++p) {
        int oidx = (((n * 16 + b2) * 8 + p) * 8) + q;
        float orr = o2r[oidx], oii = o2i[oidx];
        int e = ((((b2 * 16 + i2) * 8 + p)) << 8) + z;
        float er = E1r[e], ei = E1i[e];
        sr += orr * er - oii * ei;
        si += orr * ei + oii * er;
    }
    Ur[idx] = sr; Ui[idx] = si;
}

// dst[n,i2,z,x] += Re( sum_{b2,q} U[n,b2,i2,q,z] * E2[b2,i2,q,x] ) / 65536
__global__ __launch_bounds__(256)
void x2add_kernel(const float* __restrict__ Ur, const float* __restrict__ Ui,
                  const float* __restrict__ E2r, const float* __restrict__ E2i,
                  float* __restrict__ dst)
{
    long idx = (long)blockIdx.x * blockDim.x + threadIdx.x;   // 8,388,608
    if (idx >= (long)8388608) return;
    int x = (int)(idx & 255), z = (int)((idx >> 8) & 255);
    int i2 = (int)((idx >> 16) & 15), n = (int)(idx >> 20);
    float acc = 0.f;
    for (int b2 = 0; b2 < 16; ++b2) {
#pragma unroll
        for (int q = 0; q < 8; ++q) {
            long u = (((long)(((n * 16 + b2) * 16 + i2) * 8 + q)) << 8) + z;
            int  e = ((((b2 * 16 + i2) * 8 + q)) << 8) + x;
            acc += Ur[u] * E2r[e] - Ui[u] * E2i[e];
        }
    }
    dst[idx] += acc * (1.f / 65536.f);
}

// t[..., c, hw] += bias[c]   (HW = 65536, layout [*, C, 65536])
__global__ void bias_kernel(float* __restrict__ t, const float* __restrict__ bias,
                            int C, long total)
{
    long idx = (long)blockIdx.x * blockDim.x + threadIdx.x;
    if (idx < total) t[idx] += bias[(int)((idx >> 16) % C)];
}

// s1[c, hw] = sin(s1 + fc1_b[c]),  c in 0..127
__global__ void bias_sin_kernel(float* __restrict__ t, const float* __restrict__ bias)
{
    long idx = (long)blockIdx.x * blockDim.x + threadIdx.x;   // 128*65536
    if (idx >= (long)128 * 65536) return;
    t[idx] = sinf(t[idx] + bias[(int)(idx >> 16)]);
}

// ---------------------------------------------------------------------------
static void gemm_full(hipStream_t st, const float* A, const float* B, float* C,
                      int M, int N, int K, int lda, int ldb, int ldc,
                      long sA, long sB, long sC, float alpha, float beta, int batch)
{
    dim3 grid((unsigned)(N / 256), (unsigned)(M / 16), (unsigned)batch);
    dim3 block(32, 4, 1);
    wmma_gemm_full<<<grid, block, 0, st>>>(A, B, C, K, lda, ldb, ldc,
                                           sA, sB, sC, alpha, beta);
}

static void gemm_guard(hipStream_t st, const float* A, const float* B, float* C,
                       int M, int N, int K, int lda, int ldb, int ldc,
                       long sA, long sB, long sC,
                       float alpha, float beta, int transA, int batch)
{
    dim3 grid((unsigned)((N + 255) / 256), (unsigned)((M + 15) / 16), (unsigned)batch);
    dim3 block(32, 4, 1);
    wmma_gemm_guard<<<grid, block, 0, st>>>(A, B, C, M, N, K, lda, ldb, ldc,
                                            sA, sB, sC, alpha, beta, transA);
}

extern "C" void kernel_launch(void* const* d_in, const int* in_sizes, int n_in,
                              void* d_out, int out_size, void* d_ws, size_t ws_size,
                              hipStream_t stream)
{
    (void)in_sizes; (void)n_in; (void)out_size; (void)ws_size;

    const float* x    = (const float*)d_in[0];
    const float* fc0w = (const float*)d_in[1];
    const float* fc0b = (const float*)d_in[2];
    const float* p1   = (const float*)d_in[3];
    const float* p2   = (const float*)d_in[4];
    const float* res  = (const float*)d_in[5];
    const float* w0w  = (const float*)d_in[6];
    const float* w0b  = (const float*)d_in[7];
    const float* fc1w = (const float*)d_in[8];
    const float* fc1b = (const float*)d_in[9];
    const float* fc2w = (const float*)d_in[10];
    const float* fc2b = (const float*)d_in[11];
    const float* tx   = (const float*)d_in[12];
    const float* ty   = (const float*)d_in[13];
    float* out = (float*)d_out;

    // ---- workspace layout (f32 elements) ----
    const long IMG = 65536;                    // 256*256
    const long CH  = 8L * 16 * IMG;            // 8,388,608
    const long TAB = 16L * 16 * 8 * 256;       // 524,288
    float* p = (float*)d_ws;
    float* v   = p;  p += CH;                  // lifted activations
    float* vn  = p;  p += CH;                  // inorm(v) -> x1 -> u
    float* Ar  = p;  p += CH;                  // spectrum real
    float* Ai  = p;  p += CH;                  // spectrum imag
    float* t1r = p;  p += CH;                  // Tr -> Sr -> Pr -> s1
    float* t1i = p;  p += CH;                  // Ti -> Si -> Pi
    float* t2r = p;  p += CH;                  // Or -> Ur
    float* t2i = p;  p += CH;                  // Oi -> Ui
    float* Fr  = p;  p += IMG;
    float* Fi  = p;  p += IMG;
    float* A1r = p;  p += TAB;  float* A1i = p; p += TAB;
    float* A2r = p;  p += TAB;  float* A2i = p; p += TAB;
    float* E1r = p;  p += TAB;  float* E1i = p; p += TAB;
    float* E2r = p;  p += TAB;  float* E2i = p; p += TAB;
    float* Ttr = p;  p += TAB;  float* Tti = p; p += TAB;
    float* o2r = p;  p += 8192; float* o2i = p; p += 8192;

    // ---- precompute DFT matrix and pole tables ----
    dftmat_kernel<<<256, 256, 0, stream>>>(Fr, Fi);
    tables_kernel<<<2048, 256, 0, stream>>>(p1, p2, tx, ty,
                                            A1r, A1i, A2r, A2i, E1r, E1i, E2r, E2i);
    ttab_kernel<<<2048, 256, 0, stream>>>(res, A1r, A1i, Ttr, Tti);

    // ---- lift + instance norm ----
    lift_kernel<<<2048, 256, 0, stream>>>(x, fc0w, fc0b, v);
    inorm_kernel<<<128, 256, 0, stream>>>(v, vn, (int)IMG);

    // ---- forward 2D DFT (F symmetric => plain GEMMs), batched over 128 images
    gemm_full(stream, vn, Fr, t1r, 256, 256, 256, 256, 256, 256, IMG, 0, IMG,  1.f, 0.f, 128);
    gemm_full(stream, vn, Fi, t1i, 256, 256, 256, 256, 256, 256, IMG, 0, IMG,  1.f, 0.f, 128);
    gemm_full(stream, Fr, t1r, Ar, 256, 256, 256, 256, 256, 256, 0, IMG, IMG,  1.f, 0.f, 128);
    gemm_full(stream, Fi, t1i, Ar, 256, 256, 256, 256, 256, 256, 0, IMG, IMG, -1.f, 1.f, 128);
    gemm_full(stream, Fr, t1i, Ai, 256, 256, 256, 256, 256, 256, 0, IMG, IMG,  1.f, 0.f, 128);
    gemm_full(stream, Fi, t1r, Ai, 256, 256, 256, 256, 256, 256, 0, IMG, IMG,  1.f, 1.f, 128);

    // ---- residue path 2 (S -> out2), S reuses T buffers ----
    skernel<<<16384, 256, 0, stream>>>(Ar, Ai, A2r, A2i, t1r, t1i);
    out2_kernel<<<32, 256, 0, stream>>>(t1r, t1i, A1r, A1i, res, o2r, o2i);

    // ---- residue path 1: per-frequency channel mixing O = alpha x G ----
    okernel<<<4096, 256, 0, stream>>>(Ar, Ai, Ttr, Tti, A2r, A2i, t2r, t2i);

    // ---- inverse 2D DFT of O: conj(F)/N per dim; x1 -> vn ----
    const float invN = 1.f / 256.f;
    gemm_full(stream, t2r, Fr, t1r, 256, 256, 256, 256, 256, 256, IMG, 0, IMG,  invN, 0.f, 128);
    gemm_full(stream, t2i, Fi, t1r, 256, 256, 256, 256, 256, 256, IMG, 0, IMG,  invN, 1.f, 128);
    gemm_full(stream, t2i, Fr, t1i, 256, 256, 256, 256, 256, 256, IMG, 0, IMG,  invN, 0.f, 128);
    gemm_full(stream, t2r, Fi, t1i, 256, 256, 256, 256, 256, 256, IMG, 0, IMG, -invN, 1.f, 128);
    gemm_full(stream, Fr, t1r, vn, 256, 256, 256, 256, 256, 256, 0, IMG, IMG,  invN, 0.f, 128);
    gemm_full(stream, Fi, t1i, vn, 256, 256, 256, 256, 256, 256, 0, IMG, IMG,  invN, 1.f, 128);

    // ---- transient part: U then x2 added into vn ----
    ukernel<<<16384, 256, 0, stream>>>(o2r, o2i, E1r, E1i, t2r, t2i);
    x2add_kernel<<<32768, 256, 0, stream>>>(t2r, t2i, E2r, E2i, vn);

    // ---- inorm(x1 + x2) in place ----
    inorm_kernel<<<128, 256, 0, stream>>>(vn, vn, (int)IMG);

    // ---- 1x1 conv residual branch: vn += w0 * v (+ bias), batched over B ----
    gemm_guard(stream, w0w, v, vn, 16, (int)IMG, 16, 16, (int)IMG, (int)IMG,
               0, 16 * IMG, 16 * IMG, 1.f, 1.f, 0, 8);
    bias_kernel<<<(unsigned)((CH + 255) / 256), 256, 0, stream>>>(vn, w0b, 16, CH);

    // ---- head: per image fc1 -> sin -> fc2 ----
    for (int b = 0; b < 8; ++b) {
        const float* ub = vn + (long)b * 16 * IMG;
        float* ob = out + (long)b * 3 * IMG;
        gemm_guard(stream, fc1w, ub, t1r, 128, (int)IMG, 16, 128, (int)IMG, (int)IMG,
                   0, 0, 0, 1.f, 0.f, /*transA=*/1, 1);
        bias_sin_kernel<<<32768, 256, 0, stream>>>(t1r, fc1b);
        gemm_guard(stream, fc2w, t1r, ob, 3, (int)IMG, 128, 3, (int)IMG, (int)IMG,
                   0, 0, 0, 1.f, 0.f, /*transA=*/1, 1);
    }
    bias_kernel<<<(unsigned)((8L * 3 * IMG + 255) / 256), 256, 0, stream>>>(out, fc2b, 3, 8L * 3 * IMG);
}
